// LinearMinimumBit_51402168599335
// MI455X (gfx1250) — matrix operand order repaired and checked by reading, not compiled, not run on hardware
//
#include <hip/hip_runtime.h>
#include <hip/hip_bf16.h>

typedef __attribute__((ext_vector_type(16))) _Float16 v16h;
typedef __attribute__((ext_vector_type(8)))  _Float16 v8h;
typedef __attribute__((ext_vector_type(8)))  float    v8f;

#define IN_F     8192
#define OUT_F    8192
#define BATCH    64
#define BLOCK_N  32                 // outputs per block
#define KB       128                // K per macro-iteration (4 sub-chunks of 32)
#define NMACRO   (IN_F / KB)        // 64 macro-iterations
#define GPR      (IN_F / 16)        // 512 quant groups per output row
#define A_STRIDE 136                // 128 + 8 halfs pad -> conflict-free ds_load_b128

// Unpack 8 3-bit values from 3 byte-values (each stored in its own int32).
__device__ __forceinline__ void unpack8(int b0, int b1, int b2, int* v) {
    v[0] = (b0 >> 5) & 7;
    v[1] = (b0 >> 2) & 7;
    v[2] = ((b0 & 3) << 1) | ((b1 >> 7) & 1);
    v[3] = (b1 >> 4) & 7;
    v[4] = (b1 >> 1) & 7;
    v[5] = ((b1 & 1) << 2) | ((b2 >> 6) & 3);
    v[6] = (b2 >> 3) & 7;
    v[7] = b2 & 7;
}

__global__ __launch_bounds__(256)
void q3_linear_wmma(const float* __restrict__ x,
                    const int*   __restrict__ wq,
                    const float* __restrict__ wn,
                    const float* __restrict__ bias,
                    float*       __restrict__ out)
{
    // A tile: 64 rows x 128 K in f16, padded row stride (17.4 KB).
    __shared__ _Float16 lds_a[BATCH * A_STRIDE];
    // B fragments: [ni*4 + sub][lo/hi][lane], 16 B per entry (8 KB).
    __shared__ v8h lds_b[8][2][32];

    const int tid  = threadIdx.x;
    const int wave = tid >> 5;          // 0..7 (wave32)
    const int lane = tid & 31;
    const int mi   = wave & 3;          // M-tile 0..3 (batch rows mi*16..)
    const int ni   = wave >> 2;         // N-group 0..1
    const int n    = lane & 15;         // N within 16x16 tile
    const int h    = lane >> 4;         // K-half selector
    const int o    = blockIdx.x * BLOCK_N + ni * 16 + n;   // output feature

    // Per-lane quant-group stream: g(m) = o*512 + m*8 + mi*2 + h.
    const long   g0 = (long)o * GPR + mi * 2 + h;
    const int*   pq = wq + g0 * 6;      // 6 int32 per group (one byte each)
    const float* pn = wn + g0;

    // Cooperative A fill: thread t owns row t>>2, k-span (t&3)*32 .. +32.
    const int fr = tid >> 2;            // 0..63
    const int fk = (tid & 3) * 32;      // 0,32,64,96
    const float* xsrc = x + (long)fr * IN_F + fk;
    _Float16* adst    = &lds_a[fr * A_STRIDE + fk];

    const int fragi = ni * 4;                       // this wave's B frag base
    const int arow  = (mi * 16 + n) * A_STRIDE + h * 16;

    v8f acc = {};

    for (int m = 0; m < NMACRO; ++m) {
        // ---- cooperative A tile fill: 32 f32 -> f16 per thread ----
#pragma unroll
        for (int j = 0; j < 4; ++j) {
            float4 f0 = *(const float4*)(xsrc + j * 8);
            float4 f1 = *(const float4*)(xsrc + j * 8 + 4);
            v8h a8;
            a8[0] = (_Float16)f0.x; a8[1] = (_Float16)f0.y;
            a8[2] = (_Float16)f0.z; a8[3] = (_Float16)f0.w;
            a8[4] = (_Float16)f1.x; a8[5] = (_Float16)f1.y;
            a8[6] = (_Float16)f1.z; a8[7] = (_Float16)f1.w;
            *(v8h*)(adst + j * 8) = a8;
        }
        xsrc += KB;

        // ---- dequant: one 16-wide quant group per lane -> B frag (ni, mi) ----
        int2 p0 = *(const int2*)(pq);
        int2 p1 = *(const int2*)(pq + 2);
        int2 p2 = *(const int2*)(pq + 4);
        float norm = *pn;
        pq += 48;                       // advance 8 groups per macro-iter
        pn += 8;
        __builtin_prefetch(pq, 0, 1);   // global_prefetch_b8 for next iter

        int v[16];
        unpack8(p0.x, p0.y, p1.x, v);
        unpack8(p1.y, p2.x, p2.y, v + 8);
        const float s2 = norm * (2.0f / 7.0f);   // w = v*(2*norm/7) - norm
        const float s7 = -norm;
        v8h blo, bhi;
#pragma unroll
        for (int j = 0; j < 8; ++j) {
            blo[j] = (_Float16)__builtin_fmaf((float)v[j],     s2, s7);
            bhi[j] = (_Float16)__builtin_fmaf((float)v[8 + j], s2, s7);
        }
        lds_b[fragi + mi][0][lane] = blo;
        lds_b[fragi + mi][1][lane] = bhi;

        __syncthreads();   // A tile + all 8 B frags visible

        // ---- gather all fragment pairs first (one dscnt wait), ----
        // ---- then 4 back-to-back WMMAs into one f32 accumulator ----
        v16h af[4], bf[4];
#pragma unroll
        for (int sub = 0; sub < 4; ++sub) {
            v8h bl = lds_b[fragi + sub][0][lane];
            v8h bh = lds_b[fragi + sub][1][lane];
            bf[sub] = __builtin_shufflevector(
                bl, bh, 0, 1, 2, 3, 4, 5, 6, 7, 8, 9, 10, 11, 12, 13, 14, 15);
            // A frag in contiguous k order (same as B): element j -> K = h*16 + j
            const _Float16* ap = &lds_a[arow + sub * 32];
            af[sub] = __builtin_shufflevector(
                *(const v8h*)ap, *(const v8h*)(ap + 8),
                0, 1, 2, 3, 4, 5, 6, 7, 8, 9, 10, 11, 12, 13, 14, 15);
        }
#pragma unroll
        for (int sub = 0; sub < 4; ++sub)
            acc = __builtin_amdgcn_wmma_f32_16x16x32_f16(
                false, af[sub], false, bf[sub], (short)0, acc, false, false);

        __syncthreads();   // everyone done reading before next fill
    }

    // ---- epilogue: D layout is M = e + 8*h (within tile), N = lane%16 ----
    const float bv = bias[o];
#pragma unroll
    for (int e = 0; e < 8; ++e) {
        const int row = mi * 16 + h * 8 + e;
        out[(long)row * OUT_F + o] = acc[e] + bv;
    }
}

extern "C" void kernel_launch(void* const* d_in, const int* in_sizes, int n_in,
                              void* d_out, int out_size, void* d_ws, size_t ws_size,
                              hipStream_t stream) {
    (void)in_sizes; (void)n_in; (void)out_size; (void)d_ws; (void)ws_size;
    const float* x    = (const float*)d_in[0];
    const int*   wq   = (const int*)  d_in[1];
    const float* wn   = (const float*)d_in[2];
    const float* bias = (const float*)d_in[3];
    float*       out  = (float*)d_out;

    dim3 grid(OUT_F / BLOCK_N);   // 256 blocks: each covers 64x32 output tile
    dim3 block(256);              // 8 wave32 waves
    hipLaunchKernelGGL(q3_linear_wmma, grid, block, 0, stream,
                       x, wq, wn, bias, out);
}
